// TemporalGATClassifier_2078764172110
// MI455X (gfx1250) — matrix operand order repaired and checked by reading, not compiled, not run on hardware
//
#include <hip/hip_runtime.h>
#include <hip/hip_bf16.h>

// ---------------- constants (match reference) ----------------
#define GG   256      // graphs
#define LL   256      // nodes per graph
#define FIN  8
#define NN   (GG*LL)  // 65536
#define TCNH 32
#define GATH 48
#define HEADS 2

typedef __attribute__((ext_vector_type(16))) _Float16 v16h;
typedef __attribute__((ext_vector_type(8)))  float    v8f;

// ordered-uint encoding so unsigned atomicMax == float max
__device__ __forceinline__ unsigned fenc(float f) {
    unsigned u = __float_as_uint(f);
    return (u & 0x80000000u) ? ~u : (u | 0x80000000u);
}
__device__ __forceinline__ float fdec(unsigned u) {
    return (u & 0x80000000u) ? __uint_as_float(u & 0x7FFFFFFFu)
                             : __uint_as_float(~u);
}
#define ENC_NEG_INF 0x007FFFFFu   // fenc(-inf)

// ---------------- fill kernels (vectorized) ----------------
__global__ void fill_f32x4_kernel(float4* p, float v, int n4) {
    int i = blockIdx.x * blockDim.x + threadIdx.x;
    if (i < n4) p[i] = make_float4(v, v, v, v);
}
__global__ void fill_u32_kernel(unsigned* p, unsigned v, int n) {
    int i = blockIdx.x * blockDim.x + threadIdx.x;
    if (i < n) p[i] = v;
}

// ---------------- TCN: conv(8->32,k3) relu, conv(32->32,k3,dil2) relu, BN ----
__global__ void tcn_kernel(const float* __restrict__ x,
                           const float* __restrict__ W1, const float* __restrict__ b1,
                           const float* __restrict__ W2, const float* __restrict__ b2,
                           const float* __restrict__ g,  const float* __restrict__ bt,
                           const float* __restrict__ mu, const float* __restrict__ var,
                           float* __restrict__ hout) {
    __shared__ float shx[LL * FIN];     // 8 KB
    __shared__ float sh1[LL * TCNH];    // 32 KB
    const int gid = blockIdx.x;         // graph
    const int l   = threadIdx.x;        // node within graph
    const int n   = gid * LL + l;

    {   // vectorized stage of x row (8 floats = 2 x float4, 32B aligned)
        const float4* xr4 = (const float4*)(x + (size_t)n * FIN);
        float4 v0 = xr4[0], v1 = xr4[1];
        float4* sh4 = (float4*)(shx + l * FIN);
        sh4[0] = v0; sh4[1] = v1;
    }
    __syncthreads();

    // conv1 (kernel 3, pad 1)
    for (int o = 0; o < TCNH; ++o) {
        float acc = b1[o];
        #pragma unroll
        for (int t = 0; t < 3; ++t) {
            int ll = l + t - 1;
            if (ll >= 0 && ll < LL) {
                #pragma unroll
                for (int i = 0; i < FIN; ++i)
                    acc += W1[o * (FIN * 3) + i * 3 + t] * shx[ll * FIN + i];
            }
        }
        sh1[l * TCNH + o] = acc > 0.f ? acc : 0.f;
    }
    __syncthreads();

    // conv2 (kernel 3, dilation 2, pad 2) + relu + batchnorm
    for (int o = 0; o < TCNH; ++o) {
        float acc = b2[o];
        #pragma unroll
        for (int t = 0; t < 3; ++t) {
            int ll = l + 2 * (t - 1);
            if (ll >= 0 && ll < LL) {
                for (int i = 0; i < TCNH; ++i)
                    acc += W2[o * (TCNH * 3) + i * 3 + t] * sh1[ll * TCNH + i];
            }
        }
        acc = acc > 0.f ? acc : 0.f;
        acc = (acc - mu[o]) * rsqrtf(var[o] + 1e-5f) * g[o] + bt[o];
        hout[(size_t)n * TCNH + o] = acc;
    }
}

// ---- helpers for WMMA linear kernels ----------------------------------------
// Pack W[K,NOUT] (f32, row-major) into LDS as per-lane WMMA B fragments:
// frag index (ct*KB+kb), lane -> v16h where element i is
//   W[(kb*32 + (lane>>4)*8 + (i&7) + (i>>3)*16) * NOUT + ct*16 + (lane&15)]
template <int K_, int NOUT_>
__device__ __forceinline__ void pack_B_to_lds(const float* __restrict__ W,
                                              _Float16* __restrict__ shp) {
    constexpr int KB = K_ / 32;
    constexpr int NT = NOUT_ / 16;
    constexpr int TOT = NT * KB * 512;   // = K_ * NOUT_
    for (int idx = threadIdx.x; idx < TOT; idx += 256) {
        const int i    = idx & 15;
        const int lane = (idx >> 4) & 31;
        const int tb   = idx >> 9;           // ct*KB + kb
        const int kb   = tb % KB;
        const int ct   = tb / KB;
        const int k    = kb * 32 + (lane >> 4) * 8 + (i & 7) + ((i >> 3) << 4);
        const int col  = ct * 16 + (lane & 15);
        shp[idx] = (_Float16)W[(size_t)k * NOUT_ + col];
    }
}

// Load A strip (16 rows x K_) into per-lane WMMA fragments, ISA 16-bit layout:
// lanes 0-15 hold K {0..7, 16..23}; lanes 16-31 hold K {8..15, 24..31}
template <int K_>
__device__ __forceinline__ void load_A_frags(const float* __restrict__ xrow,
                                             int half, v16h* a) {
    constexpr int KB = K_ / 32;
    #pragma unroll
    for (int kb = 0; kb < KB; ++kb) {
        const float4 p0 = *(const float4*)(xrow + kb * 32 + half * 8);
        const float4 p1 = *(const float4*)(xrow + kb * 32 + half * 8 + 4);
        const float4 p2 = *(const float4*)(xrow + kb * 32 + half * 8 + 16);
        const float4 p3 = *(const float4*)(xrow + kb * 32 + half * 8 + 20);
        a[kb][0]  = (_Float16)p0.x; a[kb][1]  = (_Float16)p0.y;
        a[kb][2]  = (_Float16)p0.z; a[kb][3]  = (_Float16)p0.w;
        a[kb][4]  = (_Float16)p1.x; a[kb][5]  = (_Float16)p1.y;
        a[kb][6]  = (_Float16)p1.z; a[kb][7]  = (_Float16)p1.w;
        a[kb][8]  = (_Float16)p2.x; a[kb][9]  = (_Float16)p2.y;
        a[kb][10] = (_Float16)p2.z; a[kb][11] = (_Float16)p2.w;
        a[kb][12] = (_Float16)p3.x; a[kb][13] = (_Float16)p3.y;
        a[kb][14] = (_Float16)p3.z; a[kb][15] = (_Float16)p3.w;
    }
}

// ---------------- fused dual linear: Y0 = X@W0+b0, Y1 = X@W1+b1 --------------
// blockDim = 256 (8 waves); wave w owns 16-row strip (blockIdx.x*8+w), sweeps
// all column tiles with A held in registers; B fragments pre-packed in LDS so
// each lane fetches its whole fragment as one 32B LDS vector load.
template <int K_, int NOUT_>
__global__ void linear2_wmma_kernel(const float* __restrict__ X,
                                    const float* __restrict__ W0,
                                    const float* __restrict__ b0,
                                    const float* __restrict__ W1,
                                    const float* __restrict__ b1,
                                    float* __restrict__ Y0,
                                    float* __restrict__ Y1) {
    constexpr int KB = K_ / 32;
    constexpr int NT = NOUT_ / 16;
    __shared__ v16h shW0[NT * KB * 32];
    __shared__ v16h shW1[NT * KB * 32];

    pack_B_to_lds<K_, NOUT_>(W0, (_Float16*)shW0);
    pack_B_to_lds<K_, NOUT_>(W1, (_Float16*)shW1);
    __syncthreads();

    const int lane = threadIdx.x & 31;
    const int wave = threadIdx.x >> 5;
    const int half = lane >> 4;
    const int mn   = lane & 15;
    const int row0 = (blockIdx.x * 8 + wave) * 16;

    v16h a[KB];
    load_A_frags<K_>(X + (size_t)(row0 + mn) * K_, half, a);

    #pragma unroll
    for (int ct = 0; ct < NT; ++ct) {
        const int col0 = ct * 16;
        v8f c0, c1;
        {
            float bv0 = b0[col0 + mn], bv1 = b1[col0 + mn];
            #pragma unroll
            for (int j = 0; j < 8; ++j) { c0[j] = bv0; c1[j] = bv1; }
        }
        #pragma unroll
        for (int kb = 0; kb < KB; ++kb) {
            v16h f0 = shW0[(ct * KB + kb) * 32 + lane];
            v16h f1 = shW1[(ct * KB + kb) * 32 + lane];
            c0 = __builtin_amdgcn_wmma_f32_16x16x32_f16(false, a[kb], false, f0,
                                                        (short)0, c0, false, false);
            c1 = __builtin_amdgcn_wmma_f32_16x16x32_f16(false, a[kb], false, f1,
                                                        (short)0, c1, false, false);
        }
        // C/D layout: VGPR j -> M = j + half*8, N = mn
        #pragma unroll
        for (int j = 0; j < 8; ++j) {
            const size_t r = (size_t)(row0 + j + half * 8) * NOUT_ + col0 + mn;
            Y0[r] = c0[j];
            Y1[r] = c1[j];
        }
    }
}

// ---------------- single linear (classifier): Y = relu?(X@W + b) -------------
template <int K_, int NOUT_, int RELU_>
__global__ void linear_wmma_kernel(const float* __restrict__ X,
                                   const float* __restrict__ W,
                                   const float* __restrict__ bias,
                                   float* __restrict__ Y) {
    constexpr int KB = K_ / 32;
    constexpr int NT = NOUT_ / 16;
    __shared__ v16h shW[NT * KB * 32];

    pack_B_to_lds<K_, NOUT_>(W, (_Float16*)shW);
    __syncthreads();

    const int lane = threadIdx.x & 31;
    const int wave = threadIdx.x >> 5;
    const int half = lane >> 4;
    const int mn   = lane & 15;
    const int row0 = (blockIdx.x * 8 + wave) * 16;

    v16h a[KB];
    load_A_frags<K_>(X + (size_t)(row0 + mn) * K_, half, a);

    #pragma unroll
    for (int ct = 0; ct < NT; ++ct) {
        const int col0 = ct * 16;
        v8f c;
        {
            float bv = bias[col0 + mn];
            #pragma unroll
            for (int j = 0; j < 8; ++j) c[j] = bv;
        }
        #pragma unroll
        for (int kb = 0; kb < KB; ++kb) {
            v16h f = shW[(ct * KB + kb) * 32 + lane];
            c = __builtin_amdgcn_wmma_f32_16x16x32_f16(false, a[kb], false, f,
                                                       (short)0, c, false, false);
        }
        #pragma unroll
        for (int j = 0; j < 8; ++j) {
            float v = c[j];
            if (RELU_) v = v > 0.f ? v : 0.f;
            Y[(size_t)(row0 + j + half * 8) * NOUT_ + col0 + mn] = v;
        }
    }
}

// ---------------- GATv2 edge phase 1: logits + segment max -------------------
template <int H, int C>
__global__ void edge_logits_kernel(const float* __restrict__ xl,
                                   const float* __restrict__ xr,
                                   const float* __restrict__ ea,
                                   const float* __restrict__ We,   // [2, H*C]
                                   const float* __restrict__ att,  // [H, C]
                                   const int* __restrict__ src,
                                   const int* __restrict__ dst,
                                   float* __restrict__ elog,
                                   unsigned* __restrict__ nmax,
                                   int E_) {
    constexpr int HC = H * C;
    int e = blockIdx.x * blockDim.x + threadIdx.x;
    if (e >= E_) return;
    const int s = src[e], d = dst[e];
    const float a0 = ea[(size_t)e * 2 + 0], a1 = ea[(size_t)e * 2 + 1];
    #pragma unroll
    for (int h = 0; h < H; ++h) {
        float acc = 0.f;
        #pragma unroll
        for (int q = 0; q < C / 4; ++q) {
            const int ch4 = h * C + q * 4;
            float4 vl = *(const float4*)(xl + (size_t)s * HC + ch4);
            float4 vr = *(const float4*)(xr + (size_t)d * HC + ch4);
            float4 w0 = *(const float4*)(We + ch4);
            float4 w1 = *(const float4*)(We + HC + ch4);
            float4 at = *(const float4*)(att + ch4);
            float v;
            v = vl.x + vr.x + a0 * w0.x + a1 * w1.x; v = v > 0.f ? v : 0.2f * v; acc += at.x * v;
            v = vl.y + vr.y + a0 * w0.y + a1 * w1.y; v = v > 0.f ? v : 0.2f * v; acc += at.y * v;
            v = vl.z + vr.z + a0 * w0.z + a1 * w1.z; v = v > 0.f ? v : 0.2f * v; acc += at.z * v;
            v = vl.w + vr.w + a0 * w0.w + a1 * w1.w; v = v > 0.f ? v : 0.2f * v; acc += at.w * v;
        }
        elog[(size_t)e * H + h] = acc;
        atomicMax(&nmax[(size_t)d * H + h], fenc(acc));
    }
}

// ---------------- edge phase 2: exp + segment-sum denominator ----------------
__global__ void edge_expden_kernel(const int* __restrict__ dst,
                                   float* __restrict__ elog,
                                   const unsigned* __restrict__ nmax,
                                   float* __restrict__ nden,
                                   int E_, int H) {
    int e = blockIdx.x * blockDim.x + threadIdx.x;
    if (e >= E_) return;
    const int d = dst[e];
    for (int h = 0; h < H; ++h) {
        float ex = __expf(elog[(size_t)e * H + h] - fdec(nmax[(size_t)d * H + h]));
        elog[(size_t)e * H + h] = ex;
        atomicAdd(&nden[(size_t)d * H + h], ex);
    }
}

// ---------------- edge phase 3: alpha-weighted scatter-add -------------------
// one thread per (edge, 4-channel chunk): coalesced b128 gather + 4 atomics
template <int H, int C>
__global__ void edge_aggregate_kernel(const float* __restrict__ xl,
                                      const int* __restrict__ src,
                                      const int* __restrict__ dst,
                                      const float* __restrict__ elog,
                                      const float* __restrict__ nden,
                                      float* __restrict__ outacc,
                                      int E_) {
    constexpr int HC = H * C;
    constexpr int NQ = HC / 4;
    int idx = blockIdx.x * blockDim.x + threadIdx.x;
    if (idx >= E_ * NQ) return;
    const int e = idx / NQ;
    const int q = idx % NQ;
    const int h = (q * 4) / C;
    const int s = src[e], d = dst[e];
    const float alpha = elog[(size_t)e * H + h] /
                        (nden[(size_t)d * H + h] + 1e-16f);
    const float4 v = *(const float4*)(xl + (size_t)s * HC + q * 4);
    float* op = outacc + (size_t)d * HC + q * 4;
    atomicAdd(op + 0, v.x * alpha);
    atomicAdd(op + 1, v.y * alpha);
    atomicAdd(op + 2, v.z * alpha);
    atomicAdd(op + 3, v.w * alpha);
}

// ---------------- bias + ELU (in place, vectorized) --------------------------
__global__ void bias_elu_kernel(float4* __restrict__ p,
                                const float4* __restrict__ bias,
                                int total4, int HC4) {
    int i = blockIdx.x * blockDim.x + threadIdx.x;
    if (i >= total4) return;
    float4 v = p[i];
    float4 b = bias[i % HC4];
    v.x += b.x; v.y += b.y; v.z += b.z; v.w += b.w;
    v.x = v.x > 0.f ? v.x : (__expf(v.x) - 1.f);
    v.y = v.y > 0.f ? v.y : (__expf(v.y) - 1.f);
    v.z = v.z > 0.f ? v.z : (__expf(v.z) - 1.f);
    v.w = v.w > 0.f ? v.w : (__expf(v.w) - 1.f);
    p[i] = v;
}

// ---------------- per-graph mean + max pool ----------------------------------
__global__ void pool_kernel(const float* __restrict__ act,  // [N, GATH]
                            float* __restrict__ pooled) {   // [G, 2*GATH]
    const int gid = blockIdx.x;       // graph
    const int c   = threadIdx.x;      // channel (0..47)
    float sum = 0.f, mx = -3.4e38f;
    for (int l = 0; l < LL; ++l) {
        float v = act[(size_t)(gid * LL + l) * GATH + c];
        sum += v;
        mx = v > mx ? v : mx;
    }
    pooled[(size_t)gid * (2 * GATH) + c]        = sum * (1.f / (float)LL);
    pooled[(size_t)gid * (2 * GATH) + GATH + c] = mx;
}

// ---------------- final dense 128 -> 1 ---------------------------------------
__global__ void final_dot_kernel(const float* __restrict__ hid,  // [G,128]
                                 const float* __restrict__ Wc2,  // [128]
                                 const float* __restrict__ bc2,  // [1]
                                 float* __restrict__ out) {      // [G]
    int gi = blockIdx.x * blockDim.x + threadIdx.x;
    if (gi >= GG) return;
    float acc = bc2[0];
    const float4* h4 = (const float4*)(hid + (size_t)gi * 128);
    const float4* w4 = (const float4*)Wc2;
    for (int k = 0; k < 32; ++k) {
        float4 h = h4[k], w = w4[k];
        acc += h.x * w.x + h.y * w.y + h.z * w.z + h.w * w.w;
    }
    out[gi] = acc;
}

// =============================================================================
extern "C" void kernel_launch(void* const* d_in, const int* in_sizes, int n_in,
                              void* d_out, int out_size, void* d_ws, size_t ws_size,
                              hipStream_t stream) {
    (void)n_in; (void)out_size; (void)ws_size;
    const float* x      = (const float*)d_in[0];
    const int*   ei     = (const int*)d_in[1];
    const float* eattr  = (const float*)d_in[2];
    /* batch d_in[3] unused: uniform graphs */
    const float* W1     = (const float*)d_in[4];
    const float* b1     = (const float*)d_in[5];
    const float* W2     = (const float*)d_in[6];
    const float* b2     = (const float*)d_in[7];
    const float* bn_g   = (const float*)d_in[8];
    const float* bn_b   = (const float*)d_in[9];
    const float* bn_mu  = (const float*)d_in[10];
    const float* bn_var = (const float*)d_in[11];
    const float* Wl1    = (const float*)d_in[12];
    const float* bl1    = (const float*)d_in[13];
    const float* Wr1    = (const float*)d_in[14];
    const float* br1    = (const float*)d_in[15];
    const float* We1    = (const float*)d_in[16];
    const float* att1   = (const float*)d_in[17];
    const float* bias1  = (const float*)d_in[18];
    const float* Wl2    = (const float*)d_in[19];
    const float* bl2    = (const float*)d_in[20];
    const float* Wr2    = (const float*)d_in[21];
    const float* br2    = (const float*)d_in[22];
    const float* We2    = (const float*)d_in[23];
    const float* att2   = (const float*)d_in[24];
    const float* bias2  = (const float*)d_in[25];
    const float* Wc1    = (const float*)d_in[26];
    const float* bc1    = (const float*)d_in[27];
    const float* Wc2    = (const float*)d_in[28];
    const float* bc2    = (const float*)d_in[29];

    const int E_ = in_sizes[1] / 2;
    const int* src = ei;
    const int* dst = ei + E_;

    // ---- workspace layout (floats) ----
    float* ws = (float*)d_ws;
    size_t off = 0;
    float* hbuf  = ws + off;  off += (size_t)NN * TCNH;        // TCN output [N,32]
    float* xlb   = ws + off;  off += (size_t)NN * 96;          // xl (layer1 96 / layer2 48)
    float* xrb   = ws + off;  off += (size_t)NN * 96;          // xr
    float* agg1  = ws + off;  off += (size_t)NN * 96;          // layer1 accum -> act1
    float* elog  = ws + off;  off += (size_t)E_ * HEADS;       // per-edge logits/exp
    unsigned* nmax = (unsigned*)(ws + off); off += (size_t)NN * HEADS;
    float* nden  = ws + off;  off += (size_t)NN * HEADS;
    float* agg2  = ws + off;  off += (size_t)NN * GATH;        // layer2 accum -> act2
    float* pooled= ws + off;  off += (size_t)GG * 2 * GATH;
    float* hid   = ws + off;  off += (size_t)GG * 128;

    const int eb = (E_ + 255) / 256;

    // 1) TCN
    tcn_kernel<<<GG, LL, 0, stream>>>(x, W1, b1, W2, b2, bn_g, bn_b, bn_mu, bn_var, hbuf);

    // 2) layer-1 linears (K=32, Nout=96): fused WMMA, A loaded once
    linear2_wmma_kernel<32, 96><<<NN / 128, 256, 0, stream>>>(
        hbuf, Wl1, bl1, Wr1, br1, xlb, xrb);

    // 3) init softmax state + accumulator
    fill_u32_kernel<<<(NN*HEADS + 255)/256, 256, 0, stream>>>(nmax, ENC_NEG_INF, NN*HEADS);
    fill_f32x4_kernel<<<(NN*HEADS/4 + 255)/256, 256, 0, stream>>>((float4*)nden, 0.f, NN*HEADS/4);
    fill_f32x4_kernel<<<(NN*96/4 + 255)/256, 256, 0, stream>>>((float4*)agg1, 0.f, NN*96/4);

    // 4-6) GATv2 layer 1 (H=2, C=48)
    edge_logits_kernel<HEADS, GATH><<<eb, 256, 0, stream>>>(xlb, xrb, eattr, We1, att1,
                                                            src, dst, elog, nmax, E_);
    edge_expden_kernel<<<eb, 256, 0, stream>>>(dst, elog, nmax, nden, E_, HEADS);
    {
        const int nt = E_ * (HEADS * GATH / 4);
        edge_aggregate_kernel<HEADS, GATH><<<(nt + 255)/256, 256, 0, stream>>>(
            xlb, src, dst, elog, nden, agg1, E_);
    }
    // 7) + bias1, elu  -> act1 (in agg1)
    bias_elu_kernel<<<(NN*96/4 + 255)/256, 256, 0, stream>>>((float4*)agg1, (const float4*)bias1,
                                                             NN*96/4, 96/4);

    // 8) layer-2 linears (K=96, Nout=48): fused, reuse xl/xr buffers
    linear2_wmma_kernel<96, 48><<<NN / 128, 256, 0, stream>>>(
        agg1, Wl2, bl2, Wr2, br2, xlb, xrb);

    // 9) re-init softmax state (H=1) + accumulator
    fill_u32_kernel<<<(NN + 255)/256, 256, 0, stream>>>(nmax, ENC_NEG_INF, NN);
    fill_f32x4_kernel<<<(NN/4 + 255)/256, 256, 0, stream>>>((float4*)nden, 0.f, NN/4);
    fill_f32x4_kernel<<<(NN*GATH/4 + 255)/256, 256, 0, stream>>>((float4*)agg2, 0.f, NN*GATH/4);

    // 10-12) GATv2 layer 2 (H=1, C=48)
    edge_logits_kernel<1, GATH><<<eb, 256, 0, stream>>>(xlb, xrb, eattr, We2, att2,
                                                        src, dst, elog, nmax, E_);
    edge_expden_kernel<<<eb, 256, 0, stream>>>(dst, elog, nmax, nden, E_, 1);
    {
        const int nt = E_ * (GATH / 4);
        edge_aggregate_kernel<1, GATH><<<(nt + 255)/256, 256, 0, stream>>>(
            xlb, src, dst, elog, nden, agg2, E_);
    }
    // 13) + bias2, elu -> act2 (in agg2)
    bias_elu_kernel<<<(NN*GATH/4 + 255)/256, 256, 0, stream>>>((float4*)agg2, (const float4*)bias2,
                                                               NN*GATH/4, GATH/4);

    // 14) per-graph mean+max pool -> [G, 96]
    pool_kernel<<<GG, GATH, 0, stream>>>(agg2, pooled);

    // 15) classifier hidden: [256,96]@[96,128]+bc1, relu (WMMA)
    linear_wmma_kernel<96, 128, 1><<<GG / 128, 256, 0, stream>>>(pooled, Wc1, bc1, hid);

    // 16) final 128->1
    final_dot_kernel<<<1, GG, 0, stream>>>(hid, Wc2, bc2, (float*)d_out);
}
